// CVQVAECodebook_39290360824453
// MI455X (gfx1250) — matrix operand order repaired and checked by reading, not compiled, not run on hardware
//
#include <hip/hip_runtime.h>

#define TOK_D   64
#define CB_K    2048
#define CB_STRIDE 72           // 64 elems + 8 pad (16B) -> conflict-free LDS banks
#define KTILES  (CB_K / 16)    // 128 column tiles

typedef __attribute__((ext_vector_type(16))) __bf16        v16bf;
typedef __attribute__((ext_vector_type(8)))  float         v8f;
typedef __attribute__((ext_vector_type(4)))  unsigned int  u32x4;

union BfVec {
    v16bf v;
    unsigned short s[16];
    u32x4 q[2];
};

__device__ __forceinline__ unsigned short f2bf(float f) {
    unsigned int u = __float_as_uint(f);
    unsigned int r = u + 0x7FFFu + ((u >> 16) & 1u);   // round-to-nearest-even
    return (unsigned short)(r >> 16);
}

// order-preserving float -> uint mapping (monotone: a<b  <=>  key(a)<key(b))
__device__ __forceinline__ unsigned int f2ord(float f) {
    unsigned int u = __float_as_uint(f);
    return u ^ (unsigned int)(((int)u >> 31) | 0x80000000);
}

__device__ __forceinline__ void ld8bf(unsigned short* dst, const float* src) {
    float4 v0 = ((const float4*)src)[0];
    float4 v1 = ((const float4*)src)[1];
    dst[0] = f2bf(v0.x); dst[1] = f2bf(v0.y); dst[2] = f2bf(v0.z); dst[3] = f2bf(v0.w);
    dst[4] = f2bf(v1.x); dst[5] = f2bf(v1.y); dst[6] = f2bf(v1.z); dst[7] = f2bf(v1.w);
}

__global__ void vq_argmin_gather_kernel(const float* __restrict__ x,
                                        const float* __restrict__ cb,
                                        float* __restrict__ out,
                                        int ntiles) {
    extern __shared__ char smem[];
    unsigned short* cb_lds  = (unsigned short*)smem;                         // [CB_K][CB_STRIDE] bf16
    float*          esq_lds = (float*)(smem + (size_t)CB_K * CB_STRIDE * 2); // [CB_K]

    const int tid = threadIdx.x;

    // ---- stage codebook into LDS as bf16, compute ||e||^2 (fp32) ----
    for (int k = tid; k < CB_K; k += blockDim.x) {
        const float4* src = (const float4*)(cb + k * TOK_D);
        unsigned short* dst = cb_lds + k * CB_STRIDE;
        float s = 0.0f;
        #pragma unroll
        for (int i = 0; i < 16; ++i) {
            float4 v = src[i];
            s = fmaf(v.x, v.x, s); s = fmaf(v.y, v.y, s);
            s = fmaf(v.z, v.z, s); s = fmaf(v.w, v.w, s);
            dst[4 * i + 0] = f2bf(v.x); dst[4 * i + 1] = f2bf(v.y);
            dst[4 * i + 2] = f2bf(v.z); dst[4 * i + 3] = f2bf(v.w);
        }
        esq_lds[k] = s;
    }
    __syncthreads();

    const int lane  = tid & 31;
    const int li    = lane & 15;      // column / row-in-tile index
    const int h     = lane >> 4;      // wave half
    const int wave  = tid >> 5;
    const int gwave = blockIdx.x * (blockDim.x >> 5) + wave;
    const int nwav  = gridDim.x * (blockDim.x >> 5);

    for (int tile = gwave; tile < ntiles; tile += nwav) {
        const int rowbase = tile * 16;
        const float* xr = x + (size_t)(rowbase + li) * TOK_D;

        // A-matrix (16x32 bf16) per ISA layout: lane li = row, half selects K set
        BfVec a0, a1;
        ld8bf(&a0.s[0], xr + 8 * h);            // K =  0.. 7 (+8h)
        ld8bf(&a0.s[8], xr + 16 + 8 * h);       // K = 16..23 (+8h)
        ld8bf(&a1.s[0], xr + 32 + 8 * h);       // K = 32..39 (+8h)
        ld8bf(&a1.s[8], xr + 48 + 8 * h);       // K = 48..55 (+8h)

        float runmin[8];
        int   runidx[8];
        #pragma unroll
        for (int v = 0; v < 8; ++v) { runmin[v] = 3.4e38f; runidx[v] = 0; }

        // ---- double-buffered sweep over all 128 codebook column tiles ----
        BfVec b0[2], b1[2];
        float es[2];
        {
            const unsigned short* brow = cb_lds + li * CB_STRIDE + 8 * h;
            b0[0].q[0] = *(const u32x4*)(brow);
            b0[0].q[1] = *(const u32x4*)(brow + 16);
            b1[0].q[0] = *(const u32x4*)(brow + 32);
            b1[0].q[1] = *(const u32x4*)(brow + 48);
            es[0] = esq_lds[li];
        }

        #pragma unroll 2
        for (int kc = 0; kc < KTILES; ++kc) {
            const int cur = kc & 1, nxt = cur ^ 1;

            // prefetch next column tile (wraps on last iter; harmless)
            {
                const int kn = ((kc + 1) & (KTILES - 1)) * 16 + li;
                const unsigned short* brow = cb_lds + kn * CB_STRIDE + 8 * h;
                b0[nxt].q[0] = *(const u32x4*)(brow);
                b0[nxt].q[1] = *(const u32x4*)(brow + 16);
                b1[nxt].q[0] = *(const u32x4*)(brow + 32);
                b1[nxt].q[1] = *(const u32x4*)(brow + 48);
                es[nxt] = esq_lds[kn];
            }

            v8f acc = {};
            acc = __builtin_amdgcn_wmma_f32_16x16x32_bf16(
                      false, a0.v, false, b0[cur].v, (short)0, acc, false, false);
            acc = __builtin_amdgcn_wmma_f32_16x16x32_bf16(
                      false, a1.v, false, b1[cur].v, (short)0, acc, false, false);

            const int   kcol = kc * 16 + li;
            const float e    = es[cur];
            #pragma unroll
            for (int v = 0; v < 8; ++v) {
                float dist = fmaf(-2.0f, acc[v], e);   // ||x||^2 dropped (row-constant)
                bool lt = dist < runmin[v];            // strict < keeps first occurrence
                runmin[v] = lt ? dist : runmin[v];
                runidx[v] = lt ? kcol : runidx[v];
            }
        }

        // ---- branchless argmin reduction across the 16 lanes of each half ----
        // pack (ordered-dist : index) into u64; min gives argmin w/ first-index ties
        int fin[8];
        #pragma unroll
        for (int v = 0; v < 8; ++v) {
            unsigned int hi = f2ord(runmin[v]);
            unsigned int lo = (unsigned int)runidx[v];
            #pragma unroll
            for (int m = 8; m >= 1; m >>= 1) {
                unsigned int ohi = (unsigned int)__shfl_xor((int)hi, m, 32);
                unsigned int olo = (unsigned int)__shfl_xor((int)lo, m, 32);
                unsigned long long k  = ((unsigned long long)hi  << 32) | lo;
                unsigned long long ok = ((unsigned long long)ohi << 32) | olo;
                unsigned long long mn = ok < k ? ok : k;
                hi = (unsigned int)(mn >> 32);
                lo = (unsigned int)mn;
            }
            fin[v] = (int)lo;
        }

        // ---- gather winning rows from fp32 codebook, write z_q ----
        #pragma unroll
        for (int r = 0; r < 16; ++r) {
            const int idx = (r < 8) ? __shfl(fin[r], 0, 32)
                                    : __shfl(fin[r - 8], 16, 32);
            const float2* srcv = (const float2*)(cb + (size_t)idx * TOK_D);
            float2*       dstv = (float2*)(out + (size_t)(rowbase + r) * TOK_D);
            dstv[lane] = srcv[lane];
        }
    }
}

extern "C" void kernel_launch(void* const* d_in, const int* in_sizes, int n_in,
                              void* d_out, int out_size, void* d_ws, size_t ws_size,
                              hipStream_t stream) {
    const float* x  = (const float*)d_in[0];
    const float* cb = (const float*)d_in[1];
    float* out = (float*)d_out;

    const int n      = in_sizes[0] / TOK_D;   // 131072 tokens
    const int ntiles = n / 16;                // 8192 16-row tiles

    const size_t smem = (size_t)CB_K * CB_STRIDE * 2 + (size_t)CB_K * 4; // ~296 KB < 320 KB WGP LDS

    // allow >default dynamic LDS (not a stream op; safe under graph capture)
    hipFuncSetAttribute((const void*)vq_argmin_gather_kernel,
                        hipFuncAttributeMaxDynamicSharedMemorySize, (int)smem);

    const int blocks  = 256;   // 8 waves/block, 2048 waves -> 4 tiles/wave
    const int threads = 256;
    vq_argmin_gather_kernel<<<blocks, threads, smem, stream>>>(x, cb, out, ntiles);
    (void)n_in; (void)d_ws; (void)ws_size; (void)out_size;
}
